// GAT_Net_65163243815281
// MI455X (gfx1250) — compile-verified, matrix operand order checked by reading, not run on hardware
//
#include <hip/hip_runtime.h>
#include <hip/hip_bf16.h>

// ---------------------------------------------------------------------------
// GAT (primal) + GCN (dual line-graph) network for MI455X / gfx1250.
// Dense GEMMs use v_wmma_f32_16x16x32_bf16 (wave32 WMMA) with B pre-swizzled
// into the per-lane WMMA layout (packed bf16, vector loads). Segment softmax /
// GCN aggregation use order-preserving u32 atomicMax + f32 atomicAdd.
// ---------------------------------------------------------------------------

#define NPRIM   100000
#define EPRIM   1600000
#define NDUAL   1600000
#define EDUAL   3200000
#define F_IN    512
#define HEADS   8
#define HID     8
#define NCLS    16
#define NEG_SLOPE 0.2f

typedef __attribute__((ext_vector_type(16))) __bf16          v16bf;
typedef __attribute__((ext_vector_type(16))) unsigned short  v16u;
typedef __attribute__((ext_vector_type(8)))  float           v8f;

// float -> bf16 bits, round to nearest even
__device__ __forceinline__ unsigned short f2bf_bits(float f) {
    unsigned u = __float_as_uint(f);
    u += 0x7FFFu + ((u >> 16) & 1u);
    return (unsigned short)(u >> 16);
}
__device__ __forceinline__ __bf16 f2bf(float f) {
    return __builtin_bit_cast(__bf16, f2bf_bits(f));
}

// order-preserving float<->uint for atomicMax-based segment max
__device__ __forceinline__ unsigned ford(float f) {
    unsigned u = __float_as_uint(f);
    return (u & 0x80000000u) ? ~u : (u | 0x80000000u);
}
__device__ __forceinline__ float fdec(unsigned u) {
    return (u & 0x80000000u) ? __uint_as_float(u & 0x7FFFFFFFu)
                             : __uint_as_float(~u);
}

__device__ __forceinline__ float lrelu(float x) { return x > 0.f ? x : NEG_SLOPE * x; }

// ---------------------------------------------------------------------------
// Generic fill
// ---------------------------------------------------------------------------
__global__ void fill_f32(float* __restrict__ p, float v, long n) {
    long i = (long)blockIdx.x * blockDim.x + threadIdx.x;
    if (i < n) p[i] = v;
}

// ---------------------------------------------------------------------------
// Pre-swizzle B (f32 [K, NT*16] row-major) into per-lane WMMA layout, bf16.
// Flat index f = ((ks*NT + t)*32 + lane)*16 + e, so the GEMM reads one
// contiguous 32B v16bf per (kstep, tile, lane).
// ---------------------------------------------------------------------------
template <int NT>
__global__ void swizzle_B_bf16(const float* __restrict__ W,
                               unsigned short* __restrict__ Bsw, int K)
{
    int f = blockIdx.x * blockDim.x + threadIdx.x;
    int total = (K / 32) * NT * 32 * 16;
    if (f >= total) return;
    int e  = f & 15;
    int l  = (f >> 4) & 31;
    int tk = f >> 9;
    int t  = tk % NT;
    int ks = tk / NT;
    int col   = t * 16 + (l & 15);
    int khalf = l >> 4;
    int kk    = ks * 32 + khalf * 8 + ((e >> 3) << 4) + (e & 7);
    Bsw[f] = f2bf_bits(W[kk * (NT * 16) + col]);
}

// ---------------------------------------------------------------------------
// WMMA GEMM: C[M,N] = A[M,K](f32->bf16 inline) * Bsw(pre-swizzled bf16).
// One wave per 16-row tile; NT = N/16 accumulators per wave (A reused).
// Tile guard is wave-uniform -> EXEC all-ones around v_wmma.
// ---------------------------------------------------------------------------
template <int NT>
__global__ __launch_bounds__(128) void gemm_bf16_wmma(
    const float* __restrict__ A, const unsigned short* __restrict__ Bsw,
    float* __restrict__ C, int M, int K)
{
    const int N = NT * 16;
    int wave = threadIdx.x >> 5;
    int lane = threadIdx.x & 31;
    int tile = blockIdx.x * 4 + wave;
    if (tile * 16 >= M) return;                 // uniform per wave

    int mrow  = tile * 16 + (lane & 15);
    int khalf = lane >> 4;                       // 0: K 0-7/16-23, 1: K 8-15/24-31

    v8f zero = {};
    v8f acc[NT];
#pragma unroll
    for (int t = 0; t < NT; ++t) acc[t] = zero;

    const v16u* bsw = (const v16u*)Bsw;

    for (int k0 = 0; k0 < K; k0 += 32) {
        // A tile: 16-bit A layout (ISA 7.12.2): lane&15 = M, lane>>4 picks K half
        v16bf a;
        const float* ap = A + (long)mrow * K + k0 + khalf * 8;
#pragma unroll
        for (int e = 0; e < 8; ++e) a[e] = f2bf(ap[e]);
#pragma unroll
        for (int e = 0; e < 8; ++e) a[8 + e] = f2bf(ap[16 + e]);

        int ks = k0 >> 5;
#pragma unroll
        for (int t = 0; t < NT; ++t) {
            // one contiguous 32B load of pre-swizzled B (2x global_load_b128)
            v16u raw = bsw[(ks * NT + t) * 32 + lane];
            v16bf b  = __builtin_bit_cast(v16bf, raw);
            acc[t] = __builtin_amdgcn_wmma_f32_16x16x32_bf16(
                false, a, false, b, (short)0, acc[t], false, false);
        }
    }
    // C layout: VGPR r -> M = r + 8*(lane>>4); N = lane&15
#pragma unroll
    for (int t = 0; t < NT; ++t) {
        int col = t * 16 + (lane & 15);
#pragma unroll
        for (int r = 0; r < 8; ++r)
            C[(long)(tile * 16 + khalf * 8 + r) * N + col] = acc[t][r];
    }
}

// ---------------------------------------------------------------------------
// GAT: per-node attention logits  alpha_s/d[n,h] = sum_c H[n,h,c]*a[h,c]
// ---------------------------------------------------------------------------
template <int H, int C>
__global__ void gat_alpha(const float* __restrict__ Hm,
                          const float* __restrict__ aws,
                          const float* __restrict__ awd,
                          float* __restrict__ as_, float* __restrict__ ad_, int n)
{
    int i = blockIdx.x * blockDim.x + threadIdx.x;
    if (i >= n * H) return;
    int node = i / H, h = i % H;
    const float* hp = Hm + (long)node * (H * C) + h * C;
    float ss = 0.f, sd = 0.f;
#pragma unroll
    for (int c = 0; c < C; ++c) {
        float v = hp[c];
        ss += v * aws[h * C + c];
        sd += v * awd[h * C + c];
    }
    as_[i] = ss;
    ad_[i] = sd;
}

// pass 1: per-dst-per-head segment max of leaky_relu(as[src]+ad[dst])
template <int H>
__global__ void gat_edge_max(const int* __restrict__ ei, int E, int n,
                             const float* __restrict__ as_, const float* __restrict__ ad_,
                             unsigned* __restrict__ m)
{
    int idx = blockIdx.x * blockDim.x + threadIdx.x;
    if (idx >= E + n) return;
    int s, d;
    if (idx < E) { s = ei[idx]; d = ei[E + idx]; }
    else         { s = d = idx - E; }
#pragma unroll
    for (int h = 0; h < H; ++h) {
        float e = lrelu(as_[s * H + h] + ad_[d * H + h]);
        atomicMax(&m[d * H + h], ford(e));
    }
}

// pass 2: ex = exp(e - m[dst]); denom += ex; agg[dst] += ex * H[src]
template <int H, int C>
__global__ void gat_edge_sum(const int* __restrict__ ei, int E, int n,
                             const float* __restrict__ as_, const float* __restrict__ ad_,
                             const unsigned* __restrict__ m,
                             const float* __restrict__ Hm,
                             float* __restrict__ den, float* __restrict__ agg)
{
    int idx = blockIdx.x * blockDim.x + threadIdx.x;
    if (idx >= E + n) return;
    int s, d;
    if (idx < E) { s = ei[idx]; d = ei[E + idx]; }
    else         { s = d = idx - E; }
#pragma unroll
    for (int h = 0; h < H; ++h) {
        float e  = lrelu(as_[s * H + h] + ad_[d * H + h]);
        float ex = __expf(e - fdec(m[d * H + h]));
        atomicAdd(&den[d * H + h], ex);
        const float* hp = Hm + (long)s * (H * C) + h * C;
        float* op = agg + (long)d * (H * C) + h * C;
#pragma unroll
        for (int c = 0; c < C; ++c) atomicAdd(&op[c], ex * hp[c]);
    }
}

// layer-1 finalize: h2 = elu(agg/denom + b1)   (64 features, 8 heads x 8)
__global__ void gat1_finalize(const float* __restrict__ agg, const float* __restrict__ den,
                              const float* __restrict__ b, float* __restrict__ h2, int n)
{
    int i = blockIdx.x * blockDim.x + threadIdx.x;
    if (i >= n * 64) return;
    int node = i >> 6, j = i & 63, h = j >> 3;
    float v = agg[i] / (den[node * 8 + h] + 1e-16f) + b[j];
    h2[i] = v > 0.f ? v : (__expf(v) - 1.f);
}

// layer-2 finalize (heads=1, mean == identity): out = agg/denom + b2
__global__ void gat2_finalize(const float* __restrict__ agg, const float* __restrict__ den,
                              const float* __restrict__ b, float* __restrict__ out, int n)
{
    int i = blockIdx.x * blockDim.x + threadIdx.x;
    if (i >= n * NCLS) return;
    int node = i >> 4, c = i & 15;
    out[i] = agg[i] / (den[node] + 1e-16f) + b[c];
}

// ---------------------------------------------------------------------------
// Dual GCN branch
// ---------------------------------------------------------------------------
__global__ void dual_deg(const int* __restrict__ dei, float* __restrict__ deg)
{
    int i = blockIdx.x * blockDim.x + threadIdx.x;
    if (i >= EDUAL) return;
    atomicAdd(&deg[dei[EDUAL + i]], 1.0f);
}

__global__ void dual_dinv(float* __restrict__ d)   // in-place deg -> rsqrt
{
    int i = blockIdx.x * blockDim.x + threadIdx.x;
    if (i >= NDUAL) return;
    d[i] = rsqrtf(fmaxf(d[i], 1.0f));
}

// hs1[i,:] = dinv[i] * (dual_x[i,:] @ Wg1)   (4 -> 8)
__global__ void dual_h1(const float* __restrict__ dx, const float* __restrict__ Wg1,
                        const float* __restrict__ dinv, float* __restrict__ hs1)
{
    int i = blockIdx.x * blockDim.x + threadIdx.x;
    if (i >= NDUAL) return;
    float xi[4];
#pragma unroll
    for (int k = 0; k < 4; ++k) xi[k] = dx[(long)i * 4 + k];
    float di = dinv[i];
#pragma unroll
    for (int j = 0; j < 8; ++j) {
        float s = 0.f;
#pragma unroll
        for (int k = 0; k < 4; ++k) s += xi[k] * Wg1[k * 8 + j];
        hs1[(long)i * 8 + j] = di * s;
    }
}

// agg[dst,:] += hs[src,:]  over dual edges + self loops
template <int C>
__global__ void dual_agg(const int* __restrict__ dei, const float* __restrict__ hs,
                         float* __restrict__ agg)
{
    int idx = blockIdx.x * blockDim.x + threadIdx.x;
    if (idx >= EDUAL + NDUAL) return;
    int s, d;
    if (idx < EDUAL) { s = dei[idx]; d = dei[EDUAL + idx]; }
    else             { s = d = idx - EDUAL; }
    const float* hp = hs + (long)s * C;
    float* op = agg + (long)d * C;
#pragma unroll
    for (int c = 0; c < C; ++c) atomicAdd(&op[c], hp[c]);
}

// Q1 = relu(dinv*agg1 + bg1); hs2[i,:] = dinv[i]*(Q1 @ Wg2)   (8 -> 16, fused)
__global__ void dual_q1_to_hs2(const float* __restrict__ agg1, const float* __restrict__ dinv,
                               const float* __restrict__ bg1, const float* __restrict__ Wg2,
                               float* __restrict__ hs2)
{
    int i = blockIdx.x * blockDim.x + threadIdx.x;
    if (i >= NDUAL) return;
    float di = dinv[i];
    float q[8];
#pragma unroll
    for (int j = 0; j < 8; ++j)
        q[j] = fmaxf(di * agg1[(long)i * 8 + j] + bg1[j], 0.0f);
#pragma unroll
    for (int c = 0; c < 16; ++c) {
        float s = 0.f;
#pragma unroll
        for (int j = 0; j < 8; ++j) s += q[j] * Wg2[j * 16 + c];
        hs2[(long)i * 16 + c] = di * s;
    }
}

// Q = dinv[dst]*agg2 + bg2   (in-place on output region)
__global__ void dual_finalize(float* __restrict__ q, const float* __restrict__ dinv,
                              const float* __restrict__ bg2)
{
    long i = (long)blockIdx.x * blockDim.x + threadIdx.x;
    if (i >= (long)NDUAL * 16) return;
    long node = i >> 4; int c = (int)(i & 15);
    q[i] = dinv[node] * q[i] + bg2[c];
}

// ---------------------------------------------------------------------------
// Host-side orchestration
// ---------------------------------------------------------------------------
static inline int ceil_div(long a, int b) { return (int)((a + b - 1) / b); }

extern "C" void kernel_launch(void* const* d_in, const int* in_sizes, int n_in,
                              void* d_out, int out_size, void* d_ws, size_t ws_size,
                              hipStream_t stream)
{
    (void)in_sizes; (void)n_in; (void)out_size; (void)ws_size;

    const float* x    = (const float*)d_in[0];
    const int*   ei   = (const int*)  d_in[1];
    const float* dx   = (const float*)d_in[2];
    const int*   dei  = (const int*)  d_in[3];
    const float* W1   = (const float*)d_in[4];
    const float* aws1 = (const float*)d_in[5];
    const float* awd1 = (const float*)d_in[6];
    const float* b1   = (const float*)d_in[7];
    const float* W2   = (const float*)d_in[8];
    const float* aws2 = (const float*)d_in[9];
    const float* awd2 = (const float*)d_in[10];
    const float* b2   = (const float*)d_in[11];
    const float* Wg1  = (const float*)d_in[12];
    const float* bg1  = (const float*)d_in[13];
    const float* Wg2  = (const float*)d_in[14];
    const float* bg2  = (const float*)d_in[15];

    float* out  = (float*)d_out;                 // [100000, 16]
    float* qout = out + (long)NPRIM * NCLS;      // [1600000, 16]
    float* ws   = (float*)d_ws;

    const int B = 256;

    // ---------------- Dual GCN branch first (its scratch is reused later) ---
    float* dinv  = ws;                            //  1.6M
    float* hs1   = ws + 1600000L;                 // 12.8M
    float* agg1d = ws + 14400000L;                // 12.8M
    float* hs2   = ws + 27200000L;                // 25.6M   (peak ~211 MB)

    fill_f32<<<ceil_div(NDUAL, B), B, 0, stream>>>(dinv, 1.0f, NDUAL);   // self loop
    dual_deg<<<ceil_div(EDUAL, B), B, 0, stream>>>(dei, dinv);
    dual_dinv<<<ceil_div(NDUAL, B), B, 0, stream>>>(dinv);
    dual_h1<<<ceil_div(NDUAL, B), B, 0, stream>>>(dx, Wg1, dinv, hs1);
    fill_f32<<<ceil_div((long)NDUAL * 8, B), B, 0, stream>>>(agg1d, 0.f, (long)NDUAL * 8);
    dual_agg<8><<<ceil_div(EDUAL + NDUAL, B), B, 0, stream>>>(dei, hs1, agg1d);
    dual_q1_to_hs2<<<ceil_div(NDUAL, B), B, 0, stream>>>(agg1d, dinv, bg1, Wg2, hs2);
    fill_f32<<<ceil_div((long)NDUAL * 16, B), B, 0, stream>>>(qout, 0.f, (long)NDUAL * 16);
    dual_agg<16><<<ceil_div(EDUAL + NDUAL, B), B, 0, stream>>>(dei, hs2, qout);
    dual_finalize<<<ceil_div((long)NDUAL * 16, B), B, 0, stream>>>(qout, dinv, bg2);

    // ---------------- Primal GAT branch (reuses dual scratch region) -------
    float*    H1    = ws;                         //  6.4M  [N,64]
    float*    as1   = ws +  6400000L;             //  0.8M
    float*    ad1   = ws +  7200000L;             //  0.8M
    unsigned* m1    = (unsigned*)(ws + 8000000L); //  0.8M
    float*    den1  = ws +  8800000L;             //  0.8M
    float*    agg1p = ws +  9600000L;             //  6.4M
    float*    h2    = ws + 16000000L;             //  6.4M
    float*    H2    = ws + 22400000L;             //  1.6M  [N,16]
    float*    as2   = ws + 24000000L;             //  0.1M
    float*    ad2   = ws + 24100000L;
    unsigned* m2    = (unsigned*)(ws + 24200000L);
    float*    den2  = ws + 24300000L;
    float*    agg2p = ws + 24400000L;             //  1.6M
    unsigned short* Bsw1 = (unsigned short*)(ws + 26000000L);  // 32768 u16
    unsigned short* Bsw2 = (unsigned short*)(ws + 26020000L);  //  1024 u16

    const int mtiles = NPRIM / 16;                // 6250 (exact)

    // Pre-swizzle W1/W2 into WMMA B-lane layout (bf16, packed)
    swizzle_B_bf16<4><<<ceil_div(32768, B), B, 0, stream>>>(W1, Bsw1, F_IN);
    swizzle_B_bf16<1><<<ceil_div(1024, B), B, 0, stream>>>(W2, Bsw2, 64);

    // Layer 1: H1 = x @ W1  (512 -> 64), bf16 WMMA
    gemm_bf16_wmma<4><<<ceil_div(mtiles, 4), 128, 0, stream>>>(x, Bsw1, H1, NPRIM, F_IN);
    gat_alpha<HEADS, HID><<<ceil_div((long)NPRIM * HEADS, B), B, 0, stream>>>(
        H1, aws1, awd1, as1, ad1, NPRIM);
    fill_f32<<<ceil_div((long)NPRIM * 8, B), B, 0, stream>>>((float*)m1, 0.f, (long)NPRIM * 8);
    fill_f32<<<ceil_div((long)NPRIM * 8, B), B, 0, stream>>>(den1, 0.f, (long)NPRIM * 8);
    fill_f32<<<ceil_div((long)NPRIM * 64, B), B, 0, stream>>>(agg1p, 0.f, (long)NPRIM * 64);
    gat_edge_max<HEADS><<<ceil_div(EPRIM + NPRIM, B), B, 0, stream>>>(
        ei, EPRIM, NPRIM, as1, ad1, m1);
    gat_edge_sum<HEADS, HID><<<ceil_div(EPRIM + NPRIM, B), B, 0, stream>>>(
        ei, EPRIM, NPRIM, as1, ad1, m1, H1, den1, agg1p);
    gat1_finalize<<<ceil_div((long)NPRIM * 64, B), B, 0, stream>>>(agg1p, den1, b1, h2, NPRIM);

    // Layer 2: H2 = h2 @ W2  (64 -> 16), bf16 WMMA
    gemm_bf16_wmma<1><<<ceil_div(mtiles, 4), 128, 0, stream>>>(h2, Bsw2, H2, NPRIM, 64);
    gat_alpha<1, NCLS><<<ceil_div(NPRIM, B), B, 0, stream>>>(H2, aws2, awd2, as2, ad2, NPRIM);
    fill_f32<<<ceil_div(NPRIM, B), B, 0, stream>>>((float*)m2, 0.f, NPRIM);
    fill_f32<<<ceil_div(NPRIM, B), B, 0, stream>>>(den2, 0.f, NPRIM);
    fill_f32<<<ceil_div((long)NPRIM * NCLS, B), B, 0, stream>>>(agg2p, 0.f, (long)NPRIM * NCLS);
    gat_edge_max<1><<<ceil_div(EPRIM + NPRIM, B), B, 0, stream>>>(
        ei, EPRIM, NPRIM, as2, ad2, m2);
    gat_edge_sum<1, NCLS><<<ceil_div(EPRIM + NPRIM, B), B, 0, stream>>>(
        ei, EPRIM, NPRIM, as2, ad2, m2, H2, den2, agg2p);
    gat2_finalize<<<ceil_div((long)NPRIM * NCLS, B), B, 0, stream>>>(agg2p, den2, b2, out, NPRIM);
}
